// MultiHeadAttention_1726576857862
// MI455X (gfx1250) — matrix operand order repaired and checked by reading, not compiled, not run on hardware
//
#include <hip/hip_runtime.h>
#include <hip/hip_bf16.h>

// ---------------------------------------------------------------------------
// MHA forward for MI455X (gfx1250, wave32, WMMA bf16 16x16x32, f32 accum)
//   B=2, S=2048, D=1024, H=16, DH=64
// Pipeline:
//   1) convert 4 weight matrices f32 -> bf16; bit-pack mask (32 keys/word)
//   2) Q/K/V projection GEMMs (X f32 @ W bf16 + b) -> (B,H,S,DH) bf16
//      (Q additionally pre-scaled by 0.125*log2(e): softmax in log2 domain)
//   3) flash attention: online softmax via exp2, row-sums computed by an
//      extra WMMA against an all-ones fragment (no shuffle reduction for l)
//   4) out = ctx bf16 @ Wo bf16 + bo -> f32
// ---------------------------------------------------------------------------

#define BB 2
#define SS 2048
#define DD 1024
#define HH 16
#define DHH 64

typedef __attribute__((ext_vector_type(16))) __bf16 v16bf;
typedef __attribute__((ext_vector_type(8)))  float  v8f;

union Frag16 { uint4 q[2]; v16bf v; };

__device__ __forceinline__ unsigned short f32_to_bf16(float f) {
  union { float f; unsigned u; } c; c.f = f;
  unsigned u = c.u;
  u += 0x7fffu + ((u >> 16) & 1u);   // round to nearest even
  return (unsigned short)(u >> 16);
}

// ---------------------------------------------------------------------------
// Weight conversion f32 -> bf16
// ---------------------------------------------------------------------------
__global__ __launch_bounds__(256) void convert_f32_bf16(
    const float* __restrict__ in, unsigned short* __restrict__ out, int n) {
  int i = (blockIdx.x * blockDim.x + threadIdx.x) * 4;
  if (i >= n) return;
  float4 f = *(const float4*)(in + i);
  union { unsigned short s[4]; uint2 u; } t;
  t.s[0] = f32_to_bf16(f.x); t.s[1] = f32_to_bf16(f.y);
  t.s[2] = f32_to_bf16(f.z); t.s[3] = f32_to_bf16(f.w);
  *(uint2*)(out + i) = t.u;
}

// ---------------------------------------------------------------------------
// Mask bit-pack: 32 mask bytes (one key chunk) -> one uint32.
// Flat word index i == (b*S + q)*(S/32) + kc since S % 32 == 0.
// ---------------------------------------------------------------------------
__global__ __launch_bounds__(256) void pack_mask_kernel(
    const unsigned char* __restrict__ mask, unsigned* __restrict__ mpack, int nwords) {
  int i = blockIdx.x * blockDim.x + threadIdx.x;
  if (i >= nwords) return;
  const unsigned char* src = mask + (size_t)i * 32;
  uint4 a = *(const uint4*)src;
  uint4 b = *(const uint4*)(src + 16);
  unsigned arr[8] = {a.x, a.y, a.z, a.w, b.x, b.y, b.z, b.w};
  unsigned w = 0;
  #pragma unroll
  for (int g = 0; g < 8; ++g) {
    unsigned v = arr[g];
    #pragma unroll
    for (int j = 0; j < 4; ++j)
      w |= (((v >> (8 * j)) & 0xffu) ? 1u : 0u) << (g * 4 + j);
  }
  mpack[i] = w;
}

// ---------------------------------------------------------------------------
// GEMM: C[M=4096][N=1024] = (X @ W + bias) * oscale
//   IN_F32:  X is f32 (projections) else bf16 (ctx)
//   OUT_F32: write f32 (B,S,D) to d_out; else scatter bf16 to (B,H,S,DH)
// Block tile 128x128, BK=32, 256 threads = 8 waves (4 M-waves x 2 N-waves),
// each wave computes 32x64 via 2x4 WMMA 16x16 tiles.
// ---------------------------------------------------------------------------
template <bool IN_F32, bool OUT_F32>
__global__ __launch_bounds__(256) void gemm_bf16_kernel(
    const void* __restrict__ Xv, const unsigned short* __restrict__ Wb,
    const float* __restrict__ bias, void* __restrict__ Outv, float oscale) {
  // LDS: As row-major [row][k] (stride 40), Bs transposed [n][k] (stride 40)
  __shared__ __align__(16) unsigned short As[128 * 40];
  __shared__ __align__(16) unsigned short Bs[128 * 40];

  const int tid  = threadIdx.x;
  const int wid  = tid >> 5;
  const int lane = tid & 31;
  const int col  = lane & 15;        // fragment column / A row low bits
  const int half = lane >> 4;        // lane half selects K interleave
  const int bm = blockIdx.y, bn = blockIdx.x;
  const int wave_m = wid & 3, wave_n = wid >> 2;

  const float*          Xf = (const float*)Xv;
  const unsigned short* Xb = (const unsigned short*)Xv;

  v8f acc[2][4];
  for (int i = 0; i < 2; ++i)
    for (int j = 0; j < 4; ++j)
      for (int e = 0; e < 8; ++e) acc[i][j][e] = 0.0f;

  for (int k0 = 0; k0 < DD; k0 += 32) {
    __syncthreads();
    // ---- stage A tile: 128 rows x 32 k (convert f32->bf16 if needed) ----
    {
      const int row = tid >> 1;
      const int kp  = (tid & 1) * 16;
      const int grow = bm * 128 + row;
      if (IN_F32) {
        const float* src = Xf + (size_t)grow * DD + k0 + kp;
        if (k0 + 32 < DD) __builtin_prefetch(src + 32, 0, 1);  // global_prefetch_b8
        union { unsigned short s[16]; uint4 q[2]; } t;
        #pragma unroll
        for (int j = 0; j < 16; ++j) t.s[j] = f32_to_bf16(src[j]);
        *(uint4*)&As[row * 40 + kp]     = t.q[0];
        *(uint4*)&As[row * 40 + kp + 8] = t.q[1];
      } else {
        const uint4* src = (const uint4*)(Xb + (size_t)grow * DD + k0 + kp);
        if (k0 + 32 < DD) __builtin_prefetch((const void*)(src + 2), 0, 1);
        *(uint4*)&As[row * 40 + kp]     = src[0];
        *(uint4*)&As[row * 40 + kp + 8] = src[1];
      }
    }
    // ---- stage B tile transposed: Bs[n_local][k_local], 128 n x 32 k ----
    {
      const int kl = tid & 31;
      const int n0 = (tid >> 5) * 16;
      const unsigned short* src = Wb + (size_t)(k0 + kl) * DD + bn * 128 + n0;
      #pragma unroll
      for (int j = 0; j < 16; ++j) Bs[(n0 + j) * 40 + kl] = src[j];
    }
    __syncthreads();

    // ---- fragments from LDS ----
    // A: row = lane%16 (+tile), a[i]=A[row][kb+i], a[8+i]=A[row][16+kb+i], kb=half*8
    // B: col = lane%16 (+tile), b[i]=B[kb+i][col], b[8+i]=B[16+kb+i][col] (Bs^T)
    Frag16 a[2], b[4];
    const int kb  = half * 8;
    const int kb2 = half * 16;
    #pragma unroll
    for (int ms = 0; ms < 2; ++ms) {
      const int row = wave_m * 32 + ms * 16 + col;
      a[ms].q[0] = *(const uint4*)&As[row * 40 + kb];
      a[ms].q[1] = *(const uint4*)&As[row * 40 + 16 + kb];
    }
    #pragma unroll
    for (int ns = 0; ns < 4; ++ns) {
      const int c2 = wave_n * 64 + ns * 16 + col;
      b[ns].q[0] = *(const uint4*)&Bs[c2 * 40 + kb];
      b[ns].q[1] = *(const uint4*)&Bs[c2 * 40 + 16 + kb];
    }
    #pragma unroll
    for (int ms = 0; ms < 2; ++ms)
      #pragma unroll
      for (int ns = 0; ns < 4; ++ns)
        acc[ms][ns] = __builtin_amdgcn_wmma_f32_16x16x32_bf16(
            false, a[ms].v, false, b[ns].v, (short)0, acc[ms][ns], false, false);
  }

  // ---- epilogue: C layout N=lane%16, M = r + 8*half per VGPR r ----
  float* Outf          = (float*)Outv;
  unsigned short* Outb = (unsigned short*)Outv;
  #pragma unroll
  for (int ms = 0; ms < 2; ++ms) {
    const int gm_base = bm * 128 + wave_m * 32 + ms * 16 + half * 8;
    #pragma unroll
    for (int ns = 0; ns < 4; ++ns) {
      const int gn = bn * 128 + wave_n * 64 + ns * 16 + col;
      const float bv = bias[gn];
      #pragma unroll
      for (int r = 0; r < 8; ++r) {
        const int gm = gm_base + r;
        const float val = (acc[ms][ns][r] + bv) * oscale;
        if (OUT_F32) {
          Outf[(size_t)gm * DD + gn] = val;
        } else {
          const int b_ = gm >> 11, s_ = gm & (SS - 1);   // m = b*S + s
          const int h_ = gn >> 6,  dh = gn & (DHH - 1);  // n = h*DH + dh
          Outb[(((size_t)b_ * HH + h_) * SS + s_) * DHH + dh] = f32_to_bf16(val);
        }
      }
    }
  }
}

// ---------------------------------------------------------------------------
// Flash attention: grid (S/128, H, B), 256 threads = 8 waves.
// Each wave owns 16 query rows (block covers 128); loop over Sk chunks of 32:
//   scores (2 tiles x 2 WMMA, Q pre-scaled into log2 domain) -> bit-mask
//   -> online softmax (exp2; row max via shfl_xor; row sum via P @ ones WMMA)
//   -> ctx += P @ V (4 WMMA).   9 WMMA per chunk per wave.
// ---------------------------------------------------------------------------
__global__ __launch_bounds__(256) void flash_attn_kernel(
    const unsigned short* __restrict__ q_ws, const unsigned short* __restrict__ k_ws,
    const unsigned short* __restrict__ v_ws, const unsigned* __restrict__ mpack,
    unsigned short* __restrict__ ctx_ws) {
  __shared__ __align__(16) unsigned short Ks[32 * 72];    // [key][dh]
  __shared__ __align__(16) unsigned short Vt[64 * 40];    // [dh][key] (transposed)
  __shared__ __align__(16) unsigned       Mw[128];        // packed mask word per q row
  __shared__ __align__(16) unsigned short Ps[8][16 * 40]; // per-wave P [m][k]

  const int tid  = threadIdx.x;
  const int wid  = tid >> 5;
  const int lane = tid & 31;
  const int col  = lane & 15;
  const int half = lane >> 4;
  const int kb   = half * 8;
  const int kb2  = half * 16;

  const int b = blockIdx.z, h = blockIdx.y;
  const int qbase = blockIdx.x * 128;

  const unsigned short* Qp = q_ws + ((size_t)b * HH + h) * SS * DHH;
  const unsigned short* Kp = k_ws + ((size_t)b * HH + h) * SS * DHH;
  const unsigned short* Vp = v_ws + ((size_t)b * HH + h) * SS * DHH;

  // Q fragments (16 rows x 64 dh = 2 A-frags), held in registers for the loop
  Frag16 qa[2];
  {
    const unsigned short* qr = Qp + (size_t)(qbase + wid * 16 + col) * DHH;
    qa[0].q[0] = *(const uint4*)(qr + kb);
    qa[0].q[1] = *(const uint4*)(qr + 16 + kb);
    qa[1].q[0] = *(const uint4*)(qr + 32 + kb);
    qa[1].q[1] = *(const uint4*)(qr + 48 + kb);
  }

  // all-ones bf16 B fragment: D = P @ ones replicates row sums across lanes
  Frag16 ones;
  ones.q[0] = make_uint4(0x3F803F80u, 0x3F803F80u, 0x3F803F80u, 0x3F803F80u);
  ones.q[1] = ones.q[0];

  v8f acc[4], acc_l;
  for (int nt = 0; nt < 4; ++nt)
    for (int e = 0; e < 8; ++e) acc[nt][e] = 0.0f;
  for (int e = 0; e < 8; ++e) acc_l[e] = 0.0f;
  float m_run[8];
  for (int r = 0; r < 8; ++r) m_run[r] = -INFINITY;

  for (int c = 0; c < SS; c += 32) {
    __syncthreads();
    // ---- cooperative staging (256 threads) ----
    {   // K chunk: 32 keys x 64 dh, row-major; one 16B copy per thread
      const int key = tid >> 3, part = (tid & 7) * 8;
      *(uint4*)&Ks[key * 72 + part] =
          *(const uint4*)(Kp + (size_t)(c + key) * DHH + part);
    }
    {   // V chunk transposed: Vt[dh][key]; 8 elements per thread
      const int key = tid & 31, d0 = (tid >> 5) * 8;
      const unsigned short* src = Vp + (size_t)(c + key) * DHH + d0;
      #pragma unroll
      for (int j = 0; j < 8; ++j) Vt[(d0 + j) * 40 + key] = src[j];
    }
    if (tid < 128) {  // packed mask: one word (32 key bits) per q row
      Mw[tid] = mpack[((size_t)b * SS + qbase + tid) * (SS / 32) + (c >> 5)];
    }
    __syncthreads();

    // ---- scores: S = Q @ K^T  (two 16x16 N-tiles, K-dim = DH = 64) ----
    v8f sc[2];
    for (int t = 0; t < 2; ++t) {
      for (int e = 0; e < 8; ++e) sc[t][e] = 0.0f;
      #pragma unroll
      for (int d = 0; d < 2; ++d) {   // dh 0..31, 32..63
        Frag16 kf;
        kf.q[0] = *(const uint4*)&Ks[(t * 16 + col) * 72 + d * 32 + kb];
        kf.q[1] = *(const uint4*)&Ks[(t * 16 + col) * 72 + d * 32 + 16 + kb];
        sc[t] = __builtin_amdgcn_wmma_f32_16x16x32_bf16(
            false, qa[d].v, false, kf.v, (short)0, sc[t], false, false);
      }
    }
    // ---- mask (element (t,r): q row = wid*16 + r + 8*half) ----
    unsigned mrow[8];
    #pragma unroll
    for (int r = 0; r < 8; ++r) mrow[r] = Mw[wid * 16 + r + half * 8];
    #pragma unroll
    for (int t = 0; t < 2; ++t)
      #pragma unroll
      for (int r = 0; r < 8; ++r)
        if ((mrow[r] >> (t * 16 + col)) & 1u) sc[t][r] = -1e18f;

    // ---- online softmax in log2 domain (Q carries 0.125*log2e) ----
    float alpha[8];
    #pragma unroll
    for (int r = 0; r < 8; ++r) {
      float mr = fmaxf(sc[0][r], sc[1][r]);
      mr = fmaxf(mr, __shfl_xor(mr, 1, 32));
      mr = fmaxf(mr, __shfl_xor(mr, 2, 32));
      mr = fmaxf(mr, __shfl_xor(mr, 4, 32));
      mr = fmaxf(mr, __shfl_xor(mr, 8, 32));
      const float mnew = fmaxf(m_run[r], mr);
      alpha[r] = exp2f(m_run[r] - mnew);
      sc[0][r] = exp2f(sc[0][r] - mnew);
      sc[1][r] = exp2f(sc[1][r] - mnew);
      m_run[r] = mnew;
    }
    #pragma unroll
    for (int nt = 0; nt < 4; ++nt)
      #pragma unroll
      for (int r = 0; r < 8; ++r) acc[nt][r] *= alpha[r];
    #pragma unroll
    for (int r = 0; r < 8; ++r) acc_l[r] *= alpha[r];

    // ---- P: C-layout -> A-layout via per-wave LDS (in-order DS per wave) ----
    #pragma unroll
    for (int t = 0; t < 2; ++t)
      #pragma unroll
      for (int r = 0; r < 8; ++r)
        Ps[wid][(r + half * 8) * 40 + t * 16 + col] = f32_to_bf16(sc[t][r]);
    Frag16 pa;
    pa.q[0] = *(const uint4*)&Ps[wid][col * 40 + kb];
    pa.q[1] = *(const uint4*)&Ps[wid][col * 40 + 16 + kb];

    // ---- l += P @ ones (row sums, replicated over lanes) ----
    acc_l = __builtin_amdgcn_wmma_f32_16x16x32_bf16(
        false, pa.v, false, ones.v, (short)0, acc_l, false, false);

    // ---- ctx += P @ V  (4 N-tiles over DH) ----
    #pragma unroll
    for (int nt = 0; nt < 4; ++nt) {
      Frag16 vf;
      vf.q[0] = *(const uint4*)&Vt[(nt * 16 + col) * 40 + kb];
      vf.q[1] = *(const uint4*)&Vt[(nt * 16 + col) * 40 + 16 + kb];
      acc[nt] = __builtin_amdgcn_wmma_f32_16x16x32_bf16(
          false, pa.v, false, vf.v, (short)0, acc[nt], false, false);
    }
  }

  // ---- normalize and write ctx (B,S,D) bf16 ----
  #pragma unroll
  for (int nt = 0; nt < 4; ++nt)
    #pragma unroll
    for (int r = 0; r < 8; ++r) {
      const int gq = qbase + wid * 16 + r + half * 8;
      const float val = acc[nt][r] / acc_l[r];
      ctx_ws[((size_t)b * SS + gq) * DD + h * DHH + nt * 16 + col] = f32_to_bf16(val);
    }
}

// ---------------------------------------------------------------------------
// Launch
// ---------------------------------------------------------------------------
extern "C" void kernel_launch(void* const* d_in, const int* in_sizes, int n_in,
                              void* d_out, int out_size, void* d_ws, size_t ws_size,
                              hipStream_t stream) {
  const float* q  = (const float*)d_in[0];
  const float* k  = (const float*)d_in[1];
  const float* v  = (const float*)d_in[2];
  const unsigned char* mask = (const unsigned char*)d_in[3];
  const float* Wq = (const float*)d_in[4];  const float* bq = (const float*)d_in[5];
  const float* Wk = (const float*)d_in[6];  const float* bk = (const float*)d_in[7];
  const float* Wv = (const float*)d_in[8];  const float* bv = (const float*)d_in[9];
  const float* Wo = (const float*)d_in[10]; const float* bo = (const float*)d_in[11];
  float* out = (float*)d_out;

  // workspace layout (bytes): 4 bf16 weights (2MB each) + Q/K/V bf16 (8MB each)
  // + ctx bf16 (8MB) + packed mask (1MB) -> 41MB total
  char* ws = (char*)d_ws;
  const size_t WSZ = (size_t)DD * DD * 2;        // 2 MB
  const size_t ASZ = (size_t)BB * SS * DD * 2;   // 8 MB
  unsigned short* wq_b = (unsigned short*)(ws + 0 * WSZ);
  unsigned short* wk_b = (unsigned short*)(ws + 1 * WSZ);
  unsigned short* wv_b = (unsigned short*)(ws + 2 * WSZ);
  unsigned short* wo_b = (unsigned short*)(ws + 3 * WSZ);
  unsigned short* q_ws   = (unsigned short*)(ws + 4 * WSZ);
  unsigned short* k_ws   = (unsigned short*)(ws + 4 * WSZ + 1 * ASZ);
  unsigned short* v_ws   = (unsigned short*)(ws + 4 * WSZ + 2 * ASZ);
  unsigned short* ctx_ws = (unsigned short*)(ws + 4 * WSZ + 3 * ASZ);
  unsigned*       mpack  = (unsigned*)      (ws + 4 * WSZ + 4 * ASZ);

  const int nW = DD * DD;
  convert_f32_bf16<<<nW / (256 * 4), 256, 0, stream>>>(Wq, wq_b, nW);
  convert_f32_bf16<<<nW / (256 * 4), 256, 0, stream>>>(Wk, wk_b, nW);
  convert_f32_bf16<<<nW / (256 * 4), 256, 0, stream>>>(Wv, wv_b, nW);
  convert_f32_bf16<<<nW / (256 * 4), 256, 0, stream>>>(Wo, wo_b, nW);

  const int nMW = BB * SS * (SS / 32);
  pack_mask_kernel<<<(nMW + 255) / 256, 256, 0, stream>>>(mask, mpack, nMW);

  // Q pre-scaled by (1/sqrt(DH)) * log2(e) so flash softmax runs in exp2 domain
  const float qscale = 0.125f * 1.44269504088896340736f;
  dim3 ggrid(DD / 128, (BB * SS) / 128);   // (8, 32)
  gemm_bf16_kernel<true, false><<<ggrid, 256, 0, stream>>>(q, wq_b, bq, q_ws, qscale);
  gemm_bf16_kernel<true, false><<<ggrid, 256, 0, stream>>>(k, wk_b, bk, k_ws, 1.0f);
  gemm_bf16_kernel<true, false><<<ggrid, 256, 0, stream>>>(v, wv_b, bv, v_ws, 1.0f);

  flash_attn_kernel<<<dim3(SS / 128, HH, BB), 256, 0, stream>>>(q_ws, k_ws, v_ws, mpack, ctx_ws);

  gemm_bf16_kernel<false, true><<<ggrid, 256, 0, stream>>>(ctx_ws, wo_b, bo, out, 1.0f);
}
